// NeuralGraphFingerprint_40484361732769
// MI455X (gfx1250) — compile-verified, matrix operand order checked by reading, not run on hardware
//
#include <hip/hip_runtime.h>

// ---------------------------------------------------------------------------
// Neural graph fingerprint for MI455X (gfx1250, wave32, WMMA).
//  N=50000 nodes, 128 atom dim, 512 fp dim, 128 graphs, ~850000 edges.
//  Bandwidth-bound on edge scatter (~1.7 GB @ 23.3 TB/s); GEMMs (~16 GFLOP)
//  run on v_wmma_f32_16x16x32_bf16 fused with relu/softmax/graph pooling.
// ---------------------------------------------------------------------------

typedef __attribute__((ext_vector_type(16))) __bf16 v16bf;
typedef __attribute__((ext_vector_type(8)))  float  v8f;

#define ATOM_DIM 128
#define FP_DIM   512
#define MT_ROWS  64            // node rows per block
#define LPAD     8             // LDS row padding (bf16 elems)
#define MS_STRIDE (ATOM_DIM + LPAD)   // 136
#define ZS_STRIDE (FP_DIM + LPAD)     // 520
// dynamic LDS: m_s[64][136] + hn_s[64][136] + z_s[64][520], all bf16
#define MS_ELEMS  (MT_ROWS * MS_STRIDE)          // 8704
#define ZS_ELEMS  (MT_ROWS * ZS_STRIDE)          // 33280
#define SMEM_BYTES ((2 * MS_ELEMS + ZS_ELEMS) * 2)   // 101376 B < 320 KB

// ---------------------------------------------------------------------------

__global__ void ngf_zero_f32(float* __restrict__ p, int n) {
  int i = blockIdx.x * blockDim.x + threadIdx.x;
  int stride = gridDim.x * blockDim.x;
  for (; i < n; i += stride) p[i] = 0.0f;
}

// Pack a K=128 x N (row-major f32) B-matrix into WMMA-fragment-major bf16:
// tile t = nt*4 + kt (kt: 32-deep K slabs), 512 elems per tile laid out
// lane-major so each lane's 16 bf16 B-fragment elements are 32 contiguous B.
// ISA B layout (mirror of 16-bit A 16x32): lane = n%16 + 16*khi,
//   k(e) = kt*32 + 8*khi + (e<8 ? e : e+8)
__global__ void ngf_pack_b_bf16(const float* __restrict__ B,
                                __bf16* __restrict__ Bp, int ncols) {
  int idx = blockIdx.x * blockDim.x + threadIdx.x;
  int total = ncols * ATOM_DIM;
  if (idx >= total) return;
  int e    = idx & 15;
  int lane = (idx >> 4) & 31;
  int t    = idx >> 9;       // 512 elems per 16x16x32 tile
  int kt   = t & 3;          // K/32 == 4 always (K = 128)
  int nt   = t >> 2;
  int khi  = lane >> 4;
  int k    = kt * 32 + 8 * khi + (e < 8 ? e : e + 8);
  int n    = nt * 16 + (lane & 15);
  Bp[idx] = (__bf16)B[k * ncols + n];
}

// One wave per edge: coalesced float4 gather of h[src], 4 f32 atomics to m[dst].
__global__ void ngf_edge_scatter(const float4* __restrict__ h4,
                                 const int* __restrict__ src,
                                 const int* __restrict__ dst,
                                 float* __restrict__ m, int nE) {
  int wid  = (blockIdx.x * blockDim.x + threadIdx.x) >> 5;
  int lane = threadIdx.x & 31;
  if (wid >= nE) return;
  int s = src[wid];
  int d = dst[wid];
  float4 v = h4[(size_t)s * (ATOM_DIM / 4) + lane];
  float* out = m + (size_t)d * ATOM_DIM + lane * 4;
  atomicAdd(out + 0, v.x);
  atomicAdd(out + 1, v.y);
  atomicAdd(out + 2, v.z);
  atomicAdd(out + 3, v.w);
}

// --- WMMA helpers -----------------------------------------------------------

__device__ __forceinline__ v8f wmma_bf16(v16bf a, v16bf b, v8f c) {
  // (neg_a, A, neg_b, B, c_mod, C, reuse_a, reuse_b)
  return __builtin_amdgcn_wmma_f32_16x16x32_bf16(false, a, false, b,
                                                 (short)0, c, false, false);
}

// A fragment from a row-major bf16 LDS tile: lane holds row (lane&15),
// K elems {k0..k0+7, k0+16..k0+23} with k0 = kt*32 + 8*(lane>>4).
__device__ __forceinline__ v16bf load_frag_lds(const __bf16* base, int stride,
                                               int row, int k0) {
  const uint4* p0 = (const uint4*)(base + row * stride + k0);
  const uint4* p1 = (const uint4*)(base + row * stride + k0 + 16);
  union { v16bf v; uint4 q[2]; } u;
  u.q[0] = *p0;
  u.q[1] = *p1;
  return u.v;
}

// B fragment from packed global buffer: 32 contiguous bytes per lane.
__device__ __forceinline__ v16bf load_frag_g(const __bf16* __restrict__ p) {
  const uint4* q = (const uint4*)p;
  union { v16bf v; uint4 x[2]; } u;
  u.x[0] = q[0];
  u.x[1] = q[1];
  return u.v;
}

// ---------------------------------------------------------------------------
// Fused per-layer kernel: hn = relu(m @ W); z = hn @ S; phis[gid] += softmax(z).
// Block = 256 threads (8 wave32), 64 node rows. Wp/Sp are fragment-packed bf16.
// ---------------------------------------------------------------------------
__global__ __launch_bounds__(256)
void ngf_layer(const float* __restrict__ m, const __bf16* __restrict__ Wp,
               const __bf16* __restrict__ Sp, const int* __restrict__ gids,
               float* __restrict__ h_next, float* __restrict__ phis,
               int nNodes) {
  extern __shared__ __bf16 smem[];
  __bf16* m_s  = smem;                 // [64][136]
  __bf16* hn_s = smem + MS_ELEMS;      // [64][136]
  __bf16* z_s  = smem + 2 * MS_ELEMS;  // [64][520]

  const int tid  = threadIdx.x;
  const int lane = tid & 31;
  const int w    = tid >> 5;
  const int row0 = blockIdx.x * MT_ROWS;

  // Stage m tile (f32 -> bf16) into LDS; rows past N are zero.
  for (int i = tid; i < MT_ROWS * ATOM_DIM; i += 256) {
    int r = i >> 7, c = i & 127;
    int gr = row0 + r;
    float v = (gr < nNodes) ? m[(size_t)gr * ATOM_DIM + c] : 0.0f;
    m_s[r * MS_STRIDE + c] = (__bf16)v;
  }
  __syncthreads();

  const int arow  = (w & 3) * 16 + (lane & 15);
  const int khi   = (lane >> 4) * 8;
  const int ccol0 = lane & 15;
  const int crow0 = (lane >> 4) * 8;

  // ---- GEMM1: hn = relu(m @ W), per wave: 4 output tiles (mt = w&3) -------
  {
    const int mt  = w & 3;
    const int ntb = (w >> 2) * 4;
    v16bf a0 = load_frag_lds(m_s, MS_STRIDE, arow, 0 * 32 + khi);
    v16bf a1 = load_frag_lds(m_s, MS_STRIDE, arow, 1 * 32 + khi);
    v16bf a2 = load_frag_lds(m_s, MS_STRIDE, arow, 2 * 32 + khi);
    v16bf a3 = load_frag_lds(m_s, MS_STRIDE, arow, 3 * 32 + khi);
#pragma unroll
    for (int j = 0; j < 4; ++j) {
      int nt = ntb + j;
      const __bf16* bp = Wp + ((size_t)(nt * 4) * 32 + lane) * 16;
      v8f acc = {};
      acc = wmma_bf16(a0, load_frag_g(bp + 0 * 512), acc);
      acc = wmma_bf16(a1, load_frag_g(bp + 1 * 512), acc);
      acc = wmma_bf16(a2, load_frag_g(bp + 2 * 512), acc);
      acc = wmma_bf16(a3, load_frag_g(bp + 3 * 512), acc);
      int col = nt * 16 + ccol0;
#pragma unroll
      for (int r = 0; r < 8; ++r) {
        float v = acc[r];
        v = v > 0.0f ? v : 0.0f;                       // relu
        hn_s[(mt * 16 + crow0 + r) * MS_STRIDE + col] = (__bf16)v;
      }
    }
  }
  __syncthreads();

  // Persist h_next (f32) for next layer's edge gather.
  for (int i = tid; i < MT_ROWS * ATOM_DIM; i += 256) {
    int r = i >> 7, c = i & 127;
    int gr = row0 + r;
    if (gr < nNodes)
      h_next[(size_t)gr * ATOM_DIM + c] = (float)hn_s[r * MS_STRIDE + c];
  }

  // ---- GEMM2: z = hn @ S, per wave: 16 output tiles -----------------------
  {
    const int mt = w & 3;
    const int nb = (w >> 2) * 16;
    v16bf a0 = load_frag_lds(hn_s, MS_STRIDE, arow, 0 * 32 + khi);
    v16bf a1 = load_frag_lds(hn_s, MS_STRIDE, arow, 1 * 32 + khi);
    v16bf a2 = load_frag_lds(hn_s, MS_STRIDE, arow, 2 * 32 + khi);
    v16bf a3 = load_frag_lds(hn_s, MS_STRIDE, arow, 3 * 32 + khi);
    for (int j = 0; j < 16; ++j) {
      int nt = nb + j;
      const __bf16* bp = Sp + ((size_t)(nt * 4) * 32 + lane) * 16;
      v8f acc = {};
      acc = wmma_bf16(a0, load_frag_g(bp + 0 * 512), acc);
      acc = wmma_bf16(a1, load_frag_g(bp + 1 * 512), acc);
      acc = wmma_bf16(a2, load_frag_g(bp + 2 * 512), acc);
      acc = wmma_bf16(a3, load_frag_g(bp + 3 * 512), acc);
      int col = nt * 16 + ccol0;
#pragma unroll
      for (int r = 0; r < 8; ++r)
        z_s[(mt * 16 + crow0 + r) * ZS_STRIDE + col] = (__bf16)acc[r];
    }
  }
  __syncthreads();

  // ---- Row softmax + per-graph pooling (wave handles one row at a time) ---
  for (int r = w; r < MT_ROWS; r += 8) {
    int gr = row0 + r;
    if (gr >= nNodes) continue;                 // uniform across wave
    const __bf16* zr = z_s + r * ZS_STRIDE;
    float vals[16];
    float mx = -3.402823466e38f;
#pragma unroll
    for (int i = 0; i < 16; ++i) {
      float v = (float)zr[lane + 32 * i];
      vals[i] = v;
      mx = fmaxf(mx, v);
    }
#pragma unroll
    for (int o = 16; o > 0; o >>= 1) mx = fmaxf(mx, __shfl_xor(mx, o, 32));
    float s = 0.0f;
#pragma unroll
    for (int i = 0; i < 16; ++i) {
      float e = __expf(vals[i] - mx);
      vals[i] = e;
      s += e;
    }
#pragma unroll
    for (int o = 16; o > 0; o >>= 1) s += __shfl_xor(s, o, 32);
    float inv = 1.0f / s;
    int gid = gids[gr];
    float* out = phis + (size_t)gid * FP_DIM;
#pragma unroll
    for (int i = 0; i < 16; ++i)
      atomicAdd(out + lane + 32 * i, vals[i] * inv);
  }
}

// ---------------------------------------------------------------------------

extern "C" void kernel_launch(void* const* d_in, const int* in_sizes, int n_in,
                              void* d_out, int out_size, void* d_ws,
                              size_t ws_size, hipStream_t stream) {
  const float* h   = (const float*)d_in[0];
  const float* W1  = (const float*)d_in[1];
  const float* S1  = (const float*)d_in[2];
  const float* W2  = (const float*)d_in[3];
  const float* S2  = (const float*)d_in[4];
  const int*  src  = (const int*)d_in[5];
  const int*  dst  = (const int*)d_in[6];
  const int*  gids = (const int*)d_in[7];
  float* phis = (float*)d_out;

  const int N  = in_sizes[0] / ATOM_DIM;   // 50000
  const int nE = in_sizes[5];              // 850000

  // workspace layout (bytes, 256-aligned)
  char* ws = (char*)d_ws;
  size_t featBytes = (size_t)N * ATOM_DIM * sizeof(float);     // 25.6 MB
  featBytes = (featBytes + 255) & ~(size_t)255;
  float*  m  = (float*)(ws);
  float*  h1 = (float*)(ws + featBytes);
  __bf16* Wp = (__bf16*)(ws + 2 * featBytes);                  // 32 KB
  __bf16* Sp = (__bf16*)(ws + 2 * featBytes + 65536);          // 128 KB

  const int nBlocks = (N + MT_ROWS - 1) / MT_ROWS;             // 782
  const int scatterBlocks = (nE + 7) / 8;                      // wave per edge

  ngf_zero_f32<<<256, 256, 0, stream>>>(phis, out_size);

  const float* hin = h;
  const float* Ws[2] = {W1, W2};
  const float* Ss[2] = {S1, S2};
  for (int layer = 0; layer < 2; ++layer) {
    ngf_zero_f32<<<2048, 256, 0, stream>>>(m, N * ATOM_DIM);
    ngf_pack_b_bf16<<<(ATOM_DIM * ATOM_DIM) / 256, 256, 0, stream>>>(
        Ws[layer], Wp, ATOM_DIM);
    ngf_pack_b_bf16<<<(ATOM_DIM * FP_DIM) / 256, 256, 0, stream>>>(
        Ss[layer], Sp, FP_DIM);
    ngf_edge_scatter<<<scatterBlocks, 256, 0, stream>>>(
        (const float4*)hin, src, dst, m, nE);
    ngf_layer<<<nBlocks, 256, SMEM_BYTES, stream>>>(m, Wp, Sp, gids, h1, phis,
                                                    N);
    hin = h1;  // layer 2 gathers from layer-1 activations
  }
}
